// LLaMAAttention_32684701122788
// MI455X (gfx1250) — matrix operand order repaired and checked by reading, not compiled, not run on hardware
//
#include <hip/hip_runtime.h>
#include <hip/hip_bf16.h>

typedef __bf16 bf16;
typedef __attribute__((ext_vector_type(16))) __bf16 v16bf;
typedef __attribute__((ext_vector_type(8)))  float  v8f;
typedef __attribute__((ext_vector_type(4)))  unsigned int u32x4;
typedef __attribute__((ext_vector_type(8)))  int i32x8;
typedef __attribute__((ext_vector_type(4)))  int i32x4;

namespace {
constexpr int kS     = 2048;
constexpr int kHid   = 4096;
constexpr int kH     = 32;
constexpr int kHkv   = 8;
constexpr int kD     = 128;
constexpr int kKvDim = kHkv * kD;                 // 1024
constexpr float kScale = 0.08838834764831845f;    // 1/sqrt(128)
}

__device__ __forceinline__ v16bf ld16_pair(const bf16* p0, const bf16* p1) {
  v16bf r;
  ((uint4*)&r)[0] = *(const uint4*)p0;
  ((uint4*)&r)[1] = *(const uint4*)p1;
  return r;
}
__device__ __forceinline__ v16bf ld16_contig(const bf16* p) { return ld16_pair(p, p + 8); }

__device__ __forceinline__ v8f zero8() {
  v8f z = {0.f, 0.f, 0.f, 0.f, 0.f, 0.f, 0.f, 0.f};
  return z;
}

// D(16x16,f32) = A(16x32,bf16) * B(32x16,bf16) + C
__device__ __forceinline__ v8f wmma_bf16(v16bf a, v16bf b, v8f c) {
  return __builtin_amdgcn_wmma_f32_16x16x32_bf16(false, a, false, b, (short)0, c,
                                                 false, false);
}

// reductions across the 16-lane half (lanes 0-15 / 16-31), wave32
__device__ __forceinline__ float half_max(float v) {
  v = fmaxf(v, __shfl_xor(v, 1));
  v = fmaxf(v, __shfl_xor(v, 2));
  v = fmaxf(v, __shfl_xor(v, 4));
  v = fmaxf(v, __shfl_xor(v, 8));
  return v;
}
__device__ __forceinline__ float half_sum(float v) {
  v += __shfl_xor(v, 1);
  v += __shfl_xor(v, 2);
  v += __shfl_xor(v, 4);
  v += __shfl_xor(v, 8);
  return v;
}

// ---- Tensor Data Mover: 2D bf16 tile (tile_d1 rows x tile_d0 cols) -> LDS ----
// D# per CDNA5 ISA sec.8; groups 2/3 zero (2-D tensor).
// 6-arg builtin form (clang-23 / therock-10.0 headers).
__device__ __forceinline__ void tdm_load_2d(unsigned lds_addr, const void* gaddr,
                                            unsigned tensor_d0, unsigned tensor_d1,
                                            unsigned tile_d0, unsigned tile_d1,
                                            unsigned stride0_elems) {
  unsigned long long ga = (unsigned long long)(uintptr_t)gaddr;
  u32x4 g0;
  g0.x = 1u;                                             // count=1, user descriptor
  g0.y = lds_addr;                                       // LDS byte address
  g0.z = (unsigned)(ga & 0xffffffffu);                   // global_addr[31:0]
  g0.w = (unsigned)((ga >> 32) & 0x01ffffffu) | (2u << 30);  // [56:32] | type=2
  i32x8 g1;
  g1[0] = (int)(1u << 16);                               // mask=0, data_size=1 (2B)
  g1[1] = (int)((tensor_d0 & 0xffffu) << 16);            // tensor_dim0[15:0] @63:48
  g1[2] = (int)((tensor_d0 >> 16) | ((tensor_d1 & 0xffffu) << 16));
  g1[3] = (int)((tensor_d1 >> 16) | (tile_d0 << 16));    // tile_dim0 @127:112
  g1[4] = (int)tile_d1;                                  // tile_dim1 @143:128
  g1[5] = (int)stride0_elems;                            // dim0_stride[31:0]
  g1[6] = 0;
  g1[7] = 0;
  i32x4 gz4 = {0, 0, 0, 0};
  i32x8 gz8 = {0, 0, 0, 0, 0, 0, 0, 0};
  __builtin_amdgcn_tensor_load_to_lds(g0, g1, gz4, gz4, gz8, 0);
}

// ---------------- conversion / transpose helpers ----------------

__global__ void cvt_f32_bf16(const float* __restrict__ in, bf16* __restrict__ out, int n) {
  int i = blockIdx.x * blockDim.x + threadIdx.x;
  if (i < n) out[i] = (bf16)in[i];
}

// in: [Kd][Nd] fp32 row-major  ->  out: [Nd][Kd] bf16 row-major
__global__ void transpose_f32_bf16(const float* __restrict__ in, bf16* __restrict__ out,
                                   int Kd, int Nd) {
  int i = blockIdx.x * blockDim.x + threadIdx.x;
  if (i < Kd * Nd) {
    int k = i / Nd, n = i - k * Nd;
    out[(size_t)n * Kd + k] = (bf16)in[i];
  }
}

// V: [S][HKV*D] -> Vt: [HKV*D][S]
__global__ void transpose_v_bf16(const bf16* __restrict__ in, bf16* __restrict__ out) {
  int i = blockIdx.x * blockDim.x + threadIdx.x;
  if (i < kS * kKvDim) {
    int s = i / kKvDim, c = i - s * kKvDim;
    out[(size_t)c * kS + s] = in[i];
  }
}

// ---------------- tiled bf16 GEMM:  C[M,N] = A[M,K] * Bt[N,K]^T ----------------
// block: 128x128 tile, 8 waves (4x2), wave: 32x64 = 2x4 WMMA tiles, K-step 32.
// LDS staging via Tensor Data Mover (wave 0 issues 2 descriptors per K step).

template <bool OUT_BF16>
__global__ void __launch_bounds__(256)
gemm_tn_bf16(const bf16* __restrict__ A, const bf16* __restrict__ Bt,
             void* __restrict__ Cout, int M, int N, int K) {
  __shared__ __align__(16) bf16 As[128][32];
  __shared__ __align__(16) bf16 Bs[128][32];
  const int tid  = threadIdx.x;
  const int lane = tid & 31;
  const int wv   = tid >> 5;
  const int wm   = wv >> 1;          // 0..3
  const int wn   = wv & 1;           // 0..1
  const int m0   = blockIdx.y * 128;
  const int n0   = blockIdx.x * 128;
  const int ncol   = lane & 15;
  const int koff8  = (lane >> 4) * 8;
  const int koff16 = (lane >> 4) * 16;
  const unsigned ldsA = (unsigned)(uintptr_t)&As[0][0];
  const unsigned ldsB = (unsigned)(uintptr_t)&Bs[0][0];

  v8f c[2][4];
#pragma unroll
  for (int mt = 0; mt < 2; ++mt)
#pragma unroll
    for (int nt = 0; nt < 4; ++nt) c[mt][nt] = zero8();

  for (int k0 = 0; k0 < K; k0 += 32) {
    __syncthreads();
    if (wv == 0) {   // TDM: EXEC-independent wave-level DMA; TENSORcnt tracked
      tdm_load_2d(ldsA, A + (size_t)m0 * K + k0,
                  (unsigned)(K - k0), 128u, 32u, 128u, (unsigned)K);
      tdm_load_2d(ldsB, Bt + (size_t)n0 * K + k0,
                  (unsigned)(K - k0), 128u, 32u, 128u, (unsigned)K);
      __builtin_amdgcn_s_wait_tensorcnt(0);
    }
    __syncthreads();

    v16bf a[2], b[4];
#pragma unroll
    for (int mt = 0; mt < 2; ++mt) {
      const bf16* p = &As[wm * 32 + mt * 16 + ncol][koff8];
      a[mt] = ld16_pair(p, p + 16);     // A-layout: K 0-7 / +16 per half
    }
#pragma unroll
    for (int nt = 0; nt < 4; ++nt)
      b[nt] = ld16_pair(&Bs[wn * 64 + nt * 16 + ncol][koff8], &Bs[wn * 64 + nt * 16 + ncol][koff8 + 16]);  // B-layout
#pragma unroll
    for (int mt = 0; mt < 2; ++mt)
#pragma unroll
      for (int nt = 0; nt < 4; ++nt)
        c[mt][nt] = wmma_bf16(a[mt], b[nt], c[mt][nt]);
  }

  const int rhi = (lane & 16) ? 8 : 0;
#pragma unroll
  for (int mt = 0; mt < 2; ++mt)
#pragma unroll
    for (int nt = 0; nt < 4; ++nt)
#pragma unroll
      for (int r = 0; r < 8; ++r) {
        int row = m0 + wm * 32 + mt * 16 + r + rhi;
        int col = n0 + wn * 64 + nt * 16 + ncol;
        if (OUT_BF16)
          ((bf16*)Cout)[(size_t)row * N + col] = (bf16)c[mt][nt][r];
        else
          ((float*)Cout)[(size_t)row * N + col] = c[mt][nt][r];
      }
}

// ---------------- flash attention: one wave per (head, 16-row band) ----------------

__global__ void __launch_bounds__(256)
attn_kernel(const bf16* __restrict__ Qb, const bf16* __restrict__ Kb,
            const bf16* __restrict__ Vt, bf16* __restrict__ Ob) {
  __shared__ __align__(16) bf16 Ptile[8][16][64];   // per-wave P staging
  const int lane = threadIdx.x & 31;
  const int wv   = threadIdx.x >> 5;
  const int gid  = blockIdx.x * 8 + wv;
  const int head = gid >> 7;            // 0..31  (128 bands per head)
  const int band = gid & 127;
  const int m0   = band * 16;
  const int kv   = head >> 2;           // GQA: 4 query heads per kv head
  const int ncol   = lane & 15;
  const int koff8  = (lane >> 4) * 8;
  const int koff16 = (lane >> 4) * 16;
  const int rhi    = (lane & 16) ? 8 : 0;

  // Q fragments (A-layout), rows m0..m0+15, D = 128 -> 4 K-chunks of 32
  const bf16* qrow = Qb + (size_t)(m0 + ncol) * kHid + head * kD;
  v16bf qf[4];
#pragma unroll
  for (int ks = 0; ks < 4; ++ks) {
    const bf16* p = qrow + ks * 32 + koff8;
    qf[ks] = ld16_pair(p, p + 16);
  }

  float mrun[8], lrun[8];
  v8f acc[8];
#pragma unroll
  for (int r = 0; r < 8; ++r) { mrun[r] = -3.0e38f; lrun[r] = 0.f; }
#pragma unroll
  for (int dt = 0; dt < 8; ++dt) acc[dt] = zero8();

  for (int j0 = 0; j0 <= m0 + 15; j0 += 64) {
    // ---- preload all K B-fragments for this 16x64 key block (distinct regs
    //      so the scheduler can clause the loads and overlap with WMMA) ----
    v16bf kb[4][4];
#pragma unroll
    for (int nt = 0; nt < 4; ++nt) {
      const int key = j0 + nt * 16 + ncol;
      const bf16* krow = Kb + (size_t)key * kKvDim + kv * kD + koff8;
#pragma unroll
      for (int ks = 0; ks < 4; ++ks) kb[nt][ks] = ld16_pair(krow + ks * 32, krow + ks * 32 + 16);
    }
    // ---- scores S = Q K^T ----
    float sm[4][8];
#pragma unroll
    for (int nt = 0; nt < 4; ++nt) {
      v8f st = zero8();
#pragma unroll
      for (int ks = 0; ks < 4; ++ks) st = wmma_bf16(qf[ks], kb[nt][ks], st);
      const int key = j0 + nt * 16 + ncol;
#pragma unroll
      for (int r = 0; r < 8; ++r) {
        float s = st[r] * kScale;
        if (key > m0 + r + rhi) s = -1.0e9f;   // causal mask
        sm[nt][r] = s;
      }
    }
    // ---- start V B-fragment loads now; softmax VALU work hides them ----
    v16bf vb[8][2];
#pragma unroll
    for (int dt = 0; dt < 8; ++dt) {
      const bf16* vcol = Vt + (size_t)(kv * kD + dt * 16 + ncol) * kS + j0 + koff8;
#pragma unroll
      for (int ks2 = 0; ks2 < 2; ++ks2) vb[dt][ks2] = ld16_pair(vcol + ks2 * 32, vcol + ks2 * 32 + 16);
    }
    // ---- online softmax ----
    float mnew[8], alpha[8];
#pragma unroll
    for (int r = 0; r < 8; ++r) {
      float mx = fmaxf(fmaxf(sm[0][r], sm[1][r]), fmaxf(sm[2][r], sm[3][r]));
      mx = half_max(mx);
      mnew[r]  = fmaxf(mrun[r], mx);
      alpha[r] = __expf(mrun[r] - mnew[r]);
      mrun[r]  = mnew[r];
    }
    float psum[8];
#pragma unroll
    for (int r = 0; r < 8; ++r) psum[r] = 0.f;
#pragma unroll
    for (int nt = 0; nt < 4; ++nt)
#pragma unroll
      for (int r = 0; r < 8; ++r) {
        float p = __expf(sm[nt][r] - mnew[r]);
        psum[r] += p;
        Ptile[wv][r + rhi][nt * 16 + ncol] = (bf16)p;   // C-layout -> LDS
      }
#pragma unroll
    for (int r = 0; r < 8; ++r)
      lrun[r] = lrun[r] * alpha[r] + half_sum(psum[r]);
#pragma unroll
    for (int dt = 0; dt < 8; ++dt)
#pragma unroll
      for (int r = 0; r < 8; ++r) acc[dt][r] *= alpha[r];

    // wait for P stores (within-wave LDS RAW; DS ops are in-order per wave)
    asm volatile("s_wait_dscnt 0" ::: "memory");

    // reload P in A-fragment layout: 16x64 -> 2 chunks of 16x32
    v16bf pf[2];
#pragma unroll
    for (int ks2 = 0; ks2 < 2; ++ks2) {
      const bf16* p = &Ptile[wv][ncol][ks2 * 32 + koff8];
      pf[ks2] = ld16_pair(p, p + 16);
    }
    // ---- O += P V ----
#pragma unroll
    for (int dt = 0; dt < 8; ++dt)
#pragma unroll
      for (int ks2 = 0; ks2 < 2; ++ks2)
        acc[dt] = wmma_bf16(pf[ks2], vb[dt][ks2], acc[dt]);
  }

  // ---- epilogue: normalize and store attn output (bf16) ----
#pragma unroll
  for (int r = 0; r < 8; ++r) {
    float inv = 1.0f / lrun[r];
    int row = m0 + r + rhi;
#pragma unroll
    for (int dt = 0; dt < 8; ++dt)
      Ob[(size_t)row * kHid + head * kD + dt * 16 + ncol] = (bf16)(acc[dt][r] * inv);
  }
}

// ---------------- launch ----------------

extern "C" void kernel_launch(void* const* d_in, const int* in_sizes, int n_in,
                              void* d_out, int out_size, void* d_ws, size_t ws_size,
                              hipStream_t stream) {
  const float* hs = (const float*)d_in[0];
  // d_in[1] = attention_mask (causal; applied analytically in-kernel)
  const float* wq = (const float*)d_in[2];
  const float* wk = (const float*)d_in[3];
  const float* wv = (const float*)d_in[4];
  const float* wo = (const float*)d_in[5];
  (void)in_sizes; (void)n_in; (void)out_size; (void)ws_size;

  char* ws = (char*)d_ws;
  const size_t MB = 1ull << 20;
  bf16* Xb  = (bf16*)(ws + 0 * MB);     // [S][HID]        16 MB
  bf16* WqT = (bf16*)(ws + 16 * MB);    // [HID][HID]      32 MB
  bf16* WkT = (bf16*)(ws + 48 * MB);    // [KV][HID]        8 MB
  bf16* WvT = (bf16*)(ws + 56 * MB);    // [KV][HID]        8 MB
  bf16* WoT = (bf16*)(ws + 64 * MB);    // [HID][HID]      32 MB
  bf16* Qb  = (bf16*)(ws + 96 * MB);    // [S][HID]        16 MB
  bf16* Kb  = (bf16*)(ws + 112 * MB);   // [S][KV]          4 MB
  bf16* Vb  = (bf16*)(ws + 116 * MB);   // [S][KV]          4 MB
  bf16* Vt  = (bf16*)(ws + 120 * MB);   // [KV][S]          4 MB
  bf16* Ab  = (bf16*)(ws + 124 * MB);   // [S][HID]        16 MB

  cvt_f32_bf16<<<(kS * kHid) / 256, 256, 0, stream>>>(hs, Xb, kS * kHid);
  transpose_f32_bf16<<<(kHid * kHid) / 256, 256, 0, stream>>>(wq, WqT, kHid, kHid);
  transpose_f32_bf16<<<(kHid * kKvDim) / 256, 256, 0, stream>>>(wk, WkT, kHid, kKvDim);
  transpose_f32_bf16<<<(kHid * kKvDim) / 256, 256, 0, stream>>>(wv, WvT, kHid, kKvDim);
  transpose_f32_bf16<<<(kHid * kHid) / 256, 256, 0, stream>>>(wo, WoT, kHid, kHid);

  gemm_tn_bf16<true><<<dim3(kHid / 128, kS / 128), 256, 0, stream>>>(
      Xb, WqT, Qb, kS, kHid, kHid);
  gemm_tn_bf16<true><<<dim3(kKvDim / 128, kS / 128), 256, 0, stream>>>(
      Xb, WkT, Kb, kS, kKvDim, kHid);
  gemm_tn_bf16<true><<<dim3(kKvDim / 128, kS / 128), 256, 0, stream>>>(
      Xb, WvT, Vb, kS, kKvDim, kHid);

  transpose_v_bf16<<<(kS * kKvDim) / 256, 256, 0, stream>>>(Vb, Vt);

  attn_kernel<<<(kH * (kS / 16)) / 8, 256, 0, stream>>>(Qb, Kb, Vt, Ab);

  gemm_tn_bf16<false><<<dim3(kHid / 128, kS / 128), 256, 0, stream>>>(
      Ab, WoT, d_out, kS, kHid, kHid);
}
